// SeqMultiHeadAttention_8641474199858
// MI455X (gfx1250) — matrix operand order repaired and checked by reading, not compiled, not run on hardware
//
#include <hip/hip_runtime.h>
#include <hip/hip_bf16.h>

// ---------------------------------------------------------------------------
// SeqMultiHeadAttention for MI455X (gfx1250, wave32, WMMA bf16 16x16x32)
// B=8, T=1024, D=1024, H=16, Dh=64
// Pipeline: cvt(f32->bf16) -> QKV GEMM (128x128, async+double-buffered)
//           -> flash attention (async K staging) -> out GEMM + bias
// Workspace (bytes): xb 16MB | wqkvb 6MB | wob 2MB | Q/K/V 16MB ea | attnb 16MB
// ---------------------------------------------------------------------------

typedef __attribute__((ext_vector_type(16))) __bf16 v16bf;
typedef __attribute__((ext_vector_type(8)))  float  v8f;
typedef __attribute__((ext_vector_type(4)))  __bf16 v4bf;

#define WMMA_BF16(a, b, c) \
  __builtin_amdgcn_wmma_f32_16x16x32_bf16(false, (a), false, (b), (short)0, (c), false, false)

__device__ __forceinline__ v8f v8f_zero() {
  v8f z;
#pragma unroll
  for (int i = 0; i < 8; ++i) z[i] = 0.0f;
  return z;
}

// A-matrix (16x32 bf16) K offset for VGPR-pair j given lane half.
// lanes 0-15: v0..3 -> K 0..7, v4..7 -> K 16..23 ; lanes 16-31: +8
__device__ __forceinline__ int a_kbase(int j, int half) {
  return (j < 4 ? 2 * j : 8 + 2 * j) + half * 8;
}

// 16-byte async copy global -> LDS (ASYNCcnt-tracked; per-lane addresses).
__device__ __forceinline__ void async_b128(unsigned lds_addr, const void* gptr) {
  asm volatile("global_load_async_to_lds_b128 %0, %1, off"
               :: "v"(lds_addr), "v"((unsigned long long)(size_t)gptr)
               : "memory");
}
__device__ __forceinline__ void wait_async0() {
  asm volatile("s_wait_asynccnt 0x0" ::: "memory");
}
// Low 32 bits of a generic pointer to LDS == LDS byte offset (aperture layout)
__device__ __forceinline__ unsigned lds_off(const void* p) {
  return (unsigned)(size_t)p;
}

// ---------------------------------------------------------------------------
// f32 -> bf16 convert
// ---------------------------------------------------------------------------
__global__ void cvt_bf16_kernel(const float* __restrict__ in,
                                __bf16* __restrict__ out, int n) {
  int i = blockIdx.x * blockDim.x + threadIdx.x;
  int stride = gridDim.x * blockDim.x;
  for (; i < n; i += stride) out[i] = (__bf16)in[i];
}

// ---------------------------------------------------------------------------
// 128x128 bf16 GEMM mainloop: 256 threads = 8 waves (4 row-groups x 2
// col-groups). Each wave owns 32 rows x 64 cols -> 8 accumulators.
// A tile (128x32) staged by async-LDS copies; B slab (32x128) staged
// transposed via registers. Double-buffered: next A/B fetched while the
// current buffers feed the WMMAs.
// ---------------------------------------------------------------------------
__device__ __forceinline__ void gemm128x128(
    const __bf16* __restrict__ A, int lda,
    const __bf16* __restrict__ B, int ldb,
    int Kdim, int rowBase, int colBase,
    __bf16* As /*[2][128*32]*/, __bf16* Bst /*[2][128*32] transposed*/,
    v8f (&acc)[2][4]) {
  const int tid  = threadIdx.x;    // 0..255
  const int lane = tid & 31;
  const int wave = tid >> 5;       // 0..7
  const int wrow = wave >> 1;      // 0..3 : rows [wrow*32, +32)
  const int wcol = wave & 1;       // 0..1 : cols [wcol*64, +64)
  const int half = lane >> 4;
  const int m    = lane & 15;

#pragma unroll
  for (int s = 0; s < 2; ++s)
#pragma unroll
    for (int nb = 0; nb < 4; ++nb) acc[s][nb] = v8f_zero();

  const unsigned asBase = lds_off(As);

  auto issueA = [&](int buf, int k0) {
#pragma unroll
    for (int i = 0; i < 2; ++i) {
      int v = tid + i * 256;       // 0..511 : 128 rows x 4 chunks of 8 bf16
      int r = v >> 2;
      int c = (v & 3) * 8;
      unsigned lds = asBase + (unsigned)((buf * 4096 + r * 32 + c) * 2);
      async_b128(lds, &A[(size_t)(rowBase + r) * lda + k0 + c]);
    }
  };
  v4bf breg[4];
  auto loadB = [&](int k0) {
#pragma unroll
    for (int i = 0; i < 4; ++i) {
      int v = tid + i * 256;       // 0..1023 : 32 rows x 32 chunks of 4
      int r = v >> 5;
      int c = (v & 31) * 4;
      breg[i] = *(const v4bf*)&B[(size_t)(k0 + r) * ldb + colBase + c];
    }
  };
  auto storeB = [&](int buf) {
#pragma unroll
    for (int i = 0; i < 4; ++i) {
      int v = tid + i * 256;
      int r = v >> 5;
      int c = (v & 31) * 4;
#pragma unroll
      for (int e = 0; e < 4; ++e) Bst[buf * 4096 + (c + e) * 32 + r] = breg[i][e];
    }
  };

  issueA(0, 0);
  loadB(0);
  const int NIT = Kdim / 32;
  for (int it = 0; it < NIT; ++it) {
    const int buf = it & 1;
    storeB(buf);
    wait_async0();        // current A tile resident
    __syncthreads();
    if (it + 1 < NIT) {   // prefetch next while computing
      issueA(buf ^ 1, (it + 1) * 32);
      loadB((it + 1) * 32);
    }
    v16bf af[2];
#pragma unroll
    for (int s = 0; s < 2; ++s) {
      const __bf16* ar = &As[buf * 4096 + (wrow * 32 + s * 16 + m) * 32];
#pragma unroll
      for (int j = 0; j < 8; ++j) {
        int kb = a_kbase(j, half);
        af[s][2 * j]     = ar[kb];
        af[s][2 * j + 1] = ar[kb + 1];
      }
    }
#pragma unroll
    for (int nb = 0; nb < 4; ++nb) {
      v16bf bfr;
      const __bf16* br = &Bst[buf * 4096 + (wcol * 64 + nb * 16 + m) * 32];
#pragma unroll
      for (int j = 0; j < 8; ++j) {
        int kk = a_kbase(j, half);
        bfr[2 * j]     = br[kk];
        bfr[2 * j + 1] = br[kk + 1];
      }
      acc[0][nb] = WMMA_BF16(af[0], bfr, acc[0][nb]);
      acc[1][nb] = WMMA_BF16(af[1], bfr, acc[1][nb]);
    }
    __syncthreads();
  }
}

// ---------------------------------------------------------------------------
// Stage 2: QKV = X @ Wqkv + b, scattered to head-major Q/K/V bf16 buffers
// grid = (3072/128, 8192/128) = (24, 64), block = 256
// ---------------------------------------------------------------------------
__global__ __launch_bounds__(256) void qkv_gemm_kernel(
    const __bf16* __restrict__ Xb, const __bf16* __restrict__ Wb,
    const float* __restrict__ bias, __bf16* __restrict__ Qb,
    __bf16* __restrict__ Kb, __bf16* __restrict__ Vb) {
  __shared__ __align__(16) __bf16 As[2 * 128 * 32];
  __shared__ __align__(16) __bf16 Bst[2 * 128 * 32];
  v8f acc[2][4];
  const int rowBase = blockIdx.y * 128;
  const int colBase = blockIdx.x * 128;
  gemm128x128(Xb, 1024, Wb, 3072, 1024, rowBase, colBase, As, Bst, acc);

  const int lane = threadIdx.x & 31;
  const int wave = threadIdx.x >> 5;
  const int wrow = wave >> 1, wcol = wave & 1;
  const int n  = lane & 15;
  const int mb = (lane >> 4) * 8;
#pragma unroll
  for (int s = 0; s < 2; ++s) {
#pragma unroll
    for (int nb = 0; nb < 4; ++nb) {
      int c = colBase + wcol * 64 + nb * 16 + n;   // 0..3071
      float bv = bias[c];
      int sel  = c >> 10;                          // 0=q 1=k 2=v
      int dcol = c & 1023;
      int h = dcol >> 6, dh = dcol & 63;
      __bf16* dst = (sel == 0) ? Qb : ((sel == 1) ? Kb : Vb);
#pragma unroll
      for (int rr = 0; rr < 8; ++rr) {
        int r = rowBase + wrow * 32 + s * 16 + mb + rr;  // 0..8191
        int bb = r >> 10, t = r & 1023;
        float val = acc[s][nb][rr] + bv;
        dst[((size_t)(bb * 16 + h) * 1024 + t) * 64 + dh] = (__bf16)val;
      }
    }
  }
}

// ---------------------------------------------------------------------------
// Stage 3: flash attention. One block = 64 query rows of one (b,h).
// grid = 8*16*16 = 2048, block = 128 (4 waves x 16 query rows).
// K block staged by async-LDS copy; V staged transposed via registers.
// Key-masks columns >= len[b]; rows >= len[b] are zeroed at writeout.
// ---------------------------------------------------------------------------
__global__ __launch_bounds__(128) void attn_kernel(
    const __bf16* __restrict__ Qb, const __bf16* __restrict__ Kb,
    const __bf16* __restrict__ Vb, const int* __restrict__ lengths,
    __bf16* __restrict__ Ob /* [B*T, D] bf16 */) {
  __shared__ __align__(16) __bf16 Ks[64 * 64];    // [key][dh]
  __shared__ __align__(16) __bf16 Vst[64 * 64];   // [dh][key] (transposed)
  __shared__ float  Ss[64 * 64];                  // scores f32
  __shared__ __bf16 Ps[64 * 64];                  // probs bf16
  __shared__ float rowm[64], rowl[64], corr[64];

  const int tid  = threadIdx.x;
  const int lane = tid & 31;
  const int wave = tid >> 5;
  const int half = lane >> 4;
  const int m    = lane & 15;
  const int mb   = half * 8;

  const int qb = blockIdx.x & 15;
  const int h  = (blockIdx.x >> 4) & 15;
  const int b  = blockIdx.x >> 8;
  const int len = lengths[b];

  const __bf16* Qg = Qb + (size_t)(b * 16 + h) * 1024 * 64;
  const __bf16* Kg = Kb + (size_t)(b * 16 + h) * 1024 * 64;
  const __bf16* Vg = Vb + (size_t)(b * 16 + h) * 1024 * 64;
  const unsigned ksBase = lds_off(Ks);

  // Q fragments (Dh=64 -> two 16x32 A fragments), held in registers
  v16bf aq0, aq1;
  const int qrow = qb * 64 + wave * 16 + m;
#pragma unroll
  for (int j = 0; j < 8; ++j) {
    int kb = a_kbase(j, half);
    aq0[2 * j]     = Qg[qrow * 64 + kb];
    aq0[2 * j + 1] = Qg[qrow * 64 + kb + 1];
    aq1[2 * j]     = Qg[qrow * 64 + 32 + kb];
    aq1[2 * j + 1] = Qg[qrow * 64 + 32 + kb + 1];
  }

  v8f o[4];
#pragma unroll
  for (int nb = 0; nb < 4; ++nb) o[nb] = v8f_zero();
  if (tid < 64) { rowm[tid] = -3.0e38f; rowl[tid] = 0.0f; }
  __syncthreads();

  for (int kt = 0; kt < len; kt += 64) {
    // Async-stage K block (64x64 bf16 = 8KB, contiguous rows)
#pragma unroll
    for (int i = 0; i < 4; ++i) {
      int v = tid + i * 128;          // 0..511 : 64 rows x 8 chunks of 8
      int r = v >> 3;
      int c = (v & 7) * 8;
      async_b128(ksBase + (unsigned)((r * 64 + c) * 2),
                 &Kg[(size_t)(kt + r) * 64 + c]);
    }
    // Stage V transposed via registers
#pragma unroll
    for (int i = 0; i < 8; ++i) {
      int v = tid + i * 128;          // 0..1023 vec4 slots
      int r = v >> 4;                 // key 0..63
      int c = (v & 15) * 4;           // dh 0..60
      v4bf vv = *(const v4bf*)&Vg[(size_t)(kt + r) * 64 + c];
#pragma unroll
      for (int e = 0; e < 4; ++e) Vst[(c + e) * 64 + r] = vv[e];
    }
    wait_async0();
    __syncthreads();

    // S = Q @ K^T  (B fragment: B[dh][key] pairs contiguous in Ks rows)
#pragma unroll
    for (int nb = 0; nb < 4; ++nb) {
      v16bf bk0, bk1;
      const __bf16* kr = &Ks[(nb * 16 + m) * 64];
#pragma unroll
      for (int j = 0; j < 8; ++j) {
        int kk = a_kbase(j, half);
        bk0[2 * j]     = kr[kk];
        bk0[2 * j + 1] = kr[kk + 1];
        bk1[2 * j]     = kr[32 + kk];
        bk1[2 * j + 1] = kr[32 + kk + 1];
      }
      v8f s = v8f_zero();
      s = WMMA_BF16(aq0, bk0, s);
      s = WMMA_BF16(aq1, bk1, s);
      int col = kt + nb * 16 + m;
#pragma unroll
      for (int rr = 0; rr < 8; ++rr) {
        float sv = s[rr] * 0.125f;          // 1/sqrt(64)
        if (col >= len) sv = -1e9f;
        Ss[(wave * 16 + mb + rr) * 64 + nb * 16 + m] = sv;
      }
    }
    __syncthreads();

    // Online softmax: one thread per query row
    if (tid < 64) {
      float mold = rowm[tid];
      float mx = mold;
      const float* srow = &Ss[tid * 64];
      for (int c = 0; c < 64; ++c) mx = fmaxf(mx, srow[c]);
      float cf  = __expf(mold - mx);
      float sum = rowl[tid] * cf;
      __bf16* prow = &Ps[tid * 64];
      for (int c = 0; c < 64; ++c) {
        float p = __expf(srow[c] - mx);
        prow[c] = (__bf16)p;
        sum += p;
      }
      rowm[tid] = mx; rowl[tid] = sum; corr[tid] = cf;
    }
    __syncthreads();

    // O = O*corr + P @ V
    v16bf p0, p1;
#pragma unroll
    for (int j = 0; j < 8; ++j) {
      int kb = a_kbase(j, half);
      p0[2 * j]     = Ps[(wave * 16 + m) * 64 + kb];
      p0[2 * j + 1] = Ps[(wave * 16 + m) * 64 + kb + 1];
      p1[2 * j]     = Ps[(wave * 16 + m) * 64 + 32 + kb];
      p1[2 * j + 1] = Ps[(wave * 16 + m) * 64 + 32 + kb + 1];
    }
#pragma unroll
    for (int nb = 0; nb < 4; ++nb) {
#pragma unroll
      for (int rr = 0; rr < 8; ++rr) o[nb][rr] *= corr[wave * 16 + mb + rr];
      v16bf bv0, bv1;
      const __bf16* vr = &Vst[(nb * 16 + m) * 64];
#pragma unroll
      for (int j = 0; j < 8; ++j) {
        int kk = a_kbase(j, half);
        bv0[2 * j]     = vr[kk];
        bv0[2 * j + 1] = vr[kk + 1];
        bv1[2 * j]     = vr[32 + kk];
        bv1[2 * j + 1] = vr[32 + kk + 1];
      }
      o[nb] = WMMA_BF16(p0, bv0, o[nb]);
      o[nb] = WMMA_BF16(p1, bv1, o[nb]);
    }
    __syncthreads();
  }

  // Writeout: normalize, zero invalid rows, pack to [B*T, D] bf16
#pragma unroll
  for (int nb = 0; nb < 4; ++nb) {
#pragma unroll
    for (int rr = 0; rr < 8; ++rr) {
      int lrow = wave * 16 + mb + rr;
      int t = qb * 64 + lrow;
      float denom = rowl[lrow];
      float val = (t < len) ? (o[nb][rr] / denom) : 0.0f;
      Ob[((size_t)(b * 1024 + t)) * 1024 + h * 64 + nb * 16 + m] = (__bf16)val;
    }
  }
}

// ---------------------------------------------------------------------------
// Stage 4: out = attn @ Wo + b_o  (f32 output). grid = (1024/128, 8192/128)
// ---------------------------------------------------------------------------
__global__ __launch_bounds__(256) void out_gemm_kernel(
    const __bf16* __restrict__ Ab, const __bf16* __restrict__ Wb,
    const float* __restrict__ bias, float* __restrict__ out) {
  __shared__ __align__(16) __bf16 As[2 * 128 * 32];
  __shared__ __align__(16) __bf16 Bst[2 * 128 * 32];
  v8f acc[2][4];
  const int rowBase = blockIdx.y * 128;
  const int colBase = blockIdx.x * 128;
  gemm128x128(Ab, 1024, Wb, 1024, 1024, rowBase, colBase, As, Bst, acc);

  const int lane = threadIdx.x & 31;
  const int wave = threadIdx.x >> 5;
  const int wrow = wave >> 1, wcol = wave & 1;
  const int n  = lane & 15;
  const int mb = (lane >> 4) * 8;
#pragma unroll
  for (int s = 0; s < 2; ++s) {
#pragma unroll
    for (int nb = 0; nb < 4; ++nb) {
      int c = colBase + wcol * 64 + nb * 16 + n;
      float bv = bias[c];
#pragma unroll
      for (int rr = 0; rr < 8; ++rr) {
        int r = rowBase + wrow * 32 + s * 16 + mb + rr;
        out[(size_t)r * 1024 + c] = acc[s][nb][rr] + bv;
      }
    }
  }
}

// ---------------------------------------------------------------------------
extern "C" void kernel_launch(void* const* d_in, const int* in_sizes, int n_in,
                              void* d_out, int out_size, void* d_ws,
                              size_t ws_size, hipStream_t stream) {
  const float* x      = (const float*)d_in[0];
  const int*   lens   = (const int*)d_in[1];
  const float* w_qkv  = (const float*)d_in[2];
  const float* b_qkv  = (const float*)d_in[3];
  const float* w_o    = (const float*)d_in[4];
  const float* b_o    = (const float*)d_in[5];
  float* out = (float*)d_out;

  const size_t XN  = 8u * 1024u * 1024u;      // 8388608
  const size_t WQN = 1024u * 3072u;
  const size_t WON = 1024u * 1024u;

  char* ws = (char*)d_ws;
  __bf16* xb    = (__bf16*)(ws);
  __bf16* wqkvb = (__bf16*)(ws + 2 * XN);
  __bf16* wob   = (__bf16*)(ws + 2 * XN + 2 * WQN);
  __bf16* Qb    = (__bf16*)(ws + 2 * XN + 2 * WQN + 2 * WON);
  __bf16* Kb    = Qb + XN;
  __bf16* Vb    = Kb + XN;
  __bf16* attnb = Vb + XN;

  cvt_bf16_kernel<<<2048, 256, 0, stream>>>(x, xb, (int)XN);
  cvt_bf16_kernel<<<2048, 256, 0, stream>>>(w_qkv, wqkvb, (int)WQN);
  cvt_bf16_kernel<<<1024, 256, 0, stream>>>(w_o, wob, (int)WON);

  qkv_gemm_kernel<<<dim3(24, 64), 256, 0, stream>>>(xb, wqkvb, b_qkv, Qb, Kb, Vb);
  attn_kernel<<<2048, 128, 0, stream>>>(Qb, Kb, Vb, lens, attnb);
  out_gemm_kernel<<<dim3(8, 64), 256, 0, stream>>>(attnb, wob, b_o, out);
}
